// ShowAttendTellModel_51926154608716
// MI455X (gfx1250) — matrix operand order repaired and checked by reading, not compiled, not run on hardware
//
#include <hip/hip_runtime.h>
#include <hip/hip_bf16.h>
#include <math.h>

// ---------------------------------------------------------------------------
// Show-Attend-Tell forward, MI455X (gfx1250, wave32, WMMA).
// B=256, L=196, D=512, H=1024, E=512, V=10000, T=20.
// Mixed precision: fp16 WMMA (f32 accum) for all GEMMs, f32 VALU for
// softmax / tanh / LSTM state. ctx_enc kept fp16 to halve streamed bytes.
// GEMM: 256 threads (8 waves, 2x4), block tile 64x128, wave tile 32x32,
// double-buffered LDS with register staging (low per-thread pressure,
// no scratch spills).
// ---------------------------------------------------------------------------

#define B_  256
#define L_  196
#define D_  512
#define H_  1024
#define E_  512
#define T_  20

typedef _Float16 v16h __attribute__((ext_vector_type(16)));
typedef float    v8f  __attribute__((ext_vector_type(8)));

union Frag16 { v16h h; unsigned u[8]; };

__device__ __forceinline__ float sigmoidf_(float x) {
    return 1.0f / (1.0f + __expf(-x));
}

// ---------------------------------------------------------------------------
// Tiled WMMA GEMM:  C[M,N] = A[M,K] * B[N,K]^T (+ bias[n])
//   A: f32 (inline converted) or f16, row-major, lda
//   B: f16 weights, [N][K] row-major, ldb
//   C: f32 or f16, ldc
// 256 threads = 8 waves (2x4). Block tile: M=64, N=128; wave tile 32x32
// (2x2 WMMA fragments, 4 WMMAs per K-step per wave).
// Double-buffered LDS: global loads for K+1 issue before the WMMA block.
// Requirements: K % 32 == 0, M % 64 == 0, N % 128 == 0 (true at all sites).
// ---------------------------------------------------------------------------
template <bool A_F32, bool C_F16>
__global__ __launch_bounds__(256) void gemm_wmma(
    const void* __restrict__ Aptr, int lda,
    const _Float16* __restrict__ Bw, int ldb,
    void* __restrict__ Cptr, int ldc,
    const float* __restrict__ bias, int K)
{
    // padded rows: 32 halves + 8 pad -> conflict-free fragment reads
    __shared__ _Float16 As[2][64 * 40];
    __shared__ _Float16 Bs[2][128 * 40];

    const int tid  = threadIdx.x;
    const int lane = tid & 31;
    const int wv   = tid >> 5;        // 0..7
    const int wm   = wv >> 2;         // wave row (0..1) -> 32 rows
    const int wn   = wv & 3;          // wave col (0..3) -> 32 cols
    const int hi   = (lane >> 4) & 1;
    const int l15  = lane & 15;

    const int n0 = blockIdx.x * 128;
    const int m0 = blockIdx.y * 64;

    // staging registers (per thread, 256-thread block)
    float4 aR32[2];      // A f32 path: 2 chunks of 4 floats
    uint4  aR16;         // A f16 path: 1 chunk of 8 halves
    uint4  bR[2];        // B: 2 chunks of 8 halves

    auto load_tile = [&](int k0) {
        if (A_F32) {
            const float* A = (const float*)Aptr;
            #pragma unroll
            for (int i = 0; i < 2; i++) {
                int q = tid + i * 256;            // 512 chunks: row=q>>3, c4=(q&7)*4
                aR32[i] = *(const float4*)(A + (size_t)(m0 + (q >> 3)) * lda + k0 + (q & 7) * 4);
            }
        } else {
            const _Float16* A = (const _Float16*)Aptr;
            int q = tid;                          // 256 chunks: row=q>>2, c8=(q&3)*8
            aR16 = *(const uint4*)(A + (size_t)(m0 + (q >> 2)) * lda + k0 + (q & 3) * 8);
        }
        #pragma unroll
        for (int i = 0; i < 2; i++) {
            int q = tid + i * 256;                // 512 chunks: n=q>>2, c8=(q&3)*8
            bR[i] = *(const uint4*)(Bw + (size_t)(n0 + (q >> 2)) * ldb + k0 + (q & 3) * 8);
        }
    };
    auto store_tile = [&](int buf) {
        if (A_F32) {
            #pragma unroll
            for (int i = 0; i < 2; i++) {
                int q = tid + i * 256;
                _Float16* dst = &As[buf][(q >> 3) * 40 + (q & 7) * 4];
                dst[0] = (_Float16)aR32[i].x; dst[1] = (_Float16)aR32[i].y;
                dst[2] = (_Float16)aR32[i].z; dst[3] = (_Float16)aR32[i].w;
            }
        } else {
            int q = tid;
            *(uint4*)&As[buf][(q >> 2) * 40 + (q & 3) * 8] = aR16;
        }
        #pragma unroll
        for (int i = 0; i < 2; i++) {
            int q = tid + i * 256;
            *(uint4*)&Bs[buf][(q >> 2) * 40 + (q & 3) * 8] = bR[i];
        }
    };

    v8f acc[2][2] = {};

    load_tile(0);
    store_tile(0);
    int cur = 0;

    for (int k0 = 0; k0 < K; k0 += 32) {
        __syncthreads();                       // LDS[cur] visible to all waves
        const bool more = (k0 + 32 < K);
        if (more) load_tile(k0 + 32);          // global loads in flight over WMMAs

        // ---- gather fragments per documented wave32 layouts ----
        Frag16 af[2], bf[2];
        #pragma unroll
        for (int mt = 0; mt < 2; mt++) {
            const int m = wm * 32 + mt * 16 + l15;
            #pragma unroll
            for (int j = 0; j < 8; j++) {
                const int kidx = ((j < 4) ? 2 * j : 2 * j + 8) + hi * 8;
                af[mt].u[j] = *(const unsigned*)&As[cur][m * 40 + kidx];
            }
        }
        #pragma unroll
        for (int nt = 0; nt < 2; nt++) {
            const int n = wn * 32 + nt * 16 + l15;
            #pragma unroll
            for (int j = 0; j < 8; j++) {
                const int kidx = ((j < 4) ? 2 * j : 2 * j + 8) + hi * 8;
                bf[nt].u[j] = *(const unsigned*)&Bs[cur][n * 40 + kidx];
            }
        }
        #pragma unroll
        for (int mt = 0; mt < 2; mt++)
            #pragma unroll
            for (int nt = 0; nt < 2; nt++)
                acc[mt][nt] = __builtin_amdgcn_wmma_f32_16x16x32_f16(
                    false, af[mt].h, false, bf[nt].h,
                    (short)0, acc[mt][nt], false, false);

        __syncthreads();                       // all waves done reading LDS[cur]
        if (more) { store_tile(cur ^ 1); cur ^= 1; }
    }

    // ---- epilogue: f32 C layout (lane<16: N=lane, M=j; lane>=16: M=j+8) ----
    for (int mt = 0; mt < 2; mt++) {
        for (int nt = 0; nt < 2; nt++) {
            const int n = n0 + wn * 32 + nt * 16 + l15;
            const int mbase = m0 + wm * 32 + mt * 16 + hi * 8;
            const float bv = bias ? bias[n] : 0.0f;
            #pragma unroll
            for (int j = 0; j < 8; j++) {
                const float v = acc[mt][nt][j] + bv;
                if (C_F16) ((_Float16*)Cptr)[(size_t)(mbase + j) * ldc + n] = (_Float16)v;
                else       ((float*)Cptr)[(size_t)(mbase + j) * ldc + n] = v;
            }
        }
    }
}

// ---------------------------------------------------------------------------
// small helper kernels
// ---------------------------------------------------------------------------
__global__ void cvt_f32_to_f16(const float* __restrict__ in,
                               _Float16* __restrict__ out, int n) {
    int i = blockIdx.x * blockDim.x + threadIdx.x;
    if (i < n) out[i] = (_Float16)in[i];
}

// image_att_w [D][D] (k,n) row-major -> f16 [n][k]
__global__ void transpose_512_f16(const float* __restrict__ in,
                                  _Float16* __restrict__ out) {
    int i = blockIdx.x * blockDim.x + threadIdx.x;   // i = e*512 + d
    int e = i >> 9, d = i & 511;
    out[i] = (_Float16)in[d * 512 + e];
}

// fuse W_ih[4096][1024] | W_hh[4096][1024] -> Wc[4096][2048] f16
__global__ void fuse_lstm_w(const float* __restrict__ wih,
                            const float* __restrict__ whh,
                            _Float16* __restrict__ out) {
    int i = blockIdx.x * blockDim.x + threadIdx.x;   // n*2048 + k
    int n = i >> 11, k = i & 2047;
    float v = (k < 1024) ? wih[n * 1024 + k] : whh[n * 1024 + (k - 1024)];
    out[i] = (_Float16)v;
}

__global__ void fuse_bias(const float* __restrict__ a,
                          const float* __restrict__ b,
                          float* __restrict__ out) {
    int i = blockIdx.x * blockDim.x + threadIdx.x;
    if (i < 4 * H_) out[i] = a[i] + b[i];
}

// mean over L of features -> fmean [B][D] f32
__global__ void fmean_kernel(const float* __restrict__ feat,
                             float* __restrict__ fmean) {
    int b = blockIdx.x;
    for (int d = threadIdx.x; d < D_; d += blockDim.x) {
        const float* f = feat + (size_t)b * L_ * D_ + d;
        float s = 0.f;
        for (int l = 0; l < L_; l++) s += f[l * D_];
        fmean[b * D_ + d] = s * (1.0f / (float)L_);
    }
}

// h0 f32 -> xh[:, 1024:2048] f16
__global__ void h_to_xh(const float* __restrict__ h, _Float16* __restrict__ xh) {
    int i = blockIdx.x * blockDim.x + threadIdx.x;   // b*1024 + j
    int b = i >> 10, j = i & 1023;
    xh[(size_t)b * 2048 + 1024 + j] = (_Float16)h[i];
}

// embedding gather for step t -> xh[:, 0:512] f16
__global__ void embed_kernel(const float* __restrict__ emb,
                             const int* __restrict__ captions, int t,
                             _Float16* __restrict__ xh) {
    int b = blockIdx.x;
    int cap = captions[b * T_ + t];
    const float* row = emb + (size_t)cap * E_;
    for (int e = threadIdx.x; e < E_; e += blockDim.x)
        xh[(size_t)b * 2048 + e] = (_Float16)row[e];
}

// fused additive attention + softmax(mean) + context -> xh[:, 512:1024] f16
__global__ __launch_bounds__(256) void attention_kernel(
    const _Float16* __restrict__ ctx16,     // [B*L][D] f16
    const float* __restrict__ feat,         // [B*L][D] f32
    const float* __restrict__ hp,           // [B][D] f32 (h @ att_hh_w.T + b)
    const float* __restrict__ w_att,        // [D]
    _Float16* __restrict__ xh)              // [B][2048]
{
    __shared__ float hpS[D_];
    __shared__ float wS[D_];
    __shared__ float scoreS[256];
    __shared__ float red[256];

    const int b = blockIdx.x;
    const int tid = threadIdx.x;
    const int lane = tid & 31;
    const int wv = tid >> 5;

    for (int d = tid; d < D_; d += 256) {
        hpS[d] = hp[b * D_ + d];
        wS[d]  = w_att[d];
    }
    __syncthreads();

    // scores: wave-per-row over L, lane reduction over D
    const _Float16* cbase = ctx16 + (size_t)b * L_ * D_;
    for (int l = wv; l < L_; l += 8) {
        const _Float16* crow = cbase + (size_t)l * D_;
        float s = 0.f;
        for (int d = lane; d < D_; d += 32)
            s += tanhf((float)crow[d] + hpS[d]) * wS[d];
        #pragma unroll
        for (int off = 16; off > 0; off >>= 1)
            s += __shfl_xor(s, off, 32);
        if (lane == 0) scoreS[l] = s;
    }
    __syncthreads();

    // softmax over L (196), folded 1/L for the mean
    float v = (tid < L_) ? scoreS[tid] : -INFINITY;
    red[tid] = v; __syncthreads();
    for (int s = 128; s > 0; s >>= 1) {
        if (tid < s) red[tid] = fmaxf(red[tid], red[tid + s]);
        __syncthreads();
    }
    const float mx = red[0]; __syncthreads();
    float e = (tid < L_) ? __expf(v - mx) : 0.0f;
    red[tid] = e; __syncthreads();
    for (int s = 128; s > 0; s >>= 1) {
        if (tid < s) red[tid] += red[tid + s];
        __syncthreads();
    }
    const float inv = 1.0f / (red[0] * (float)L_);
    __syncthreads();
    if (tid < L_) scoreS[tid] = e * inv;
    __syncthreads();

    // context[b][d] = sum_l feat[b][l][d] * alpha[l] / L   -> xh f16
    const float* fbase = feat + (size_t)b * L_ * D_;
    for (int d = tid; d < D_; d += 256) {
        float acc = 0.f;
        const float* f = fbase + d;
        for (int l = 0; l < L_; l++) acc += f[l * D_] * scoreS[l];
        xh[(size_t)b * 2048 + 512 + d] = (_Float16)acc;
    }
}

// LSTM pointwise: gates [B][4096] -> c,h; writes masked output + h f16 to xh
__global__ void lstm_kernel(const float* __restrict__ gates,
                            float* __restrict__ c,
                            const int* __restrict__ masks, int t,
                            _Float16* __restrict__ xh,
                            float* __restrict__ out) {
    int idx = blockIdx.x * blockDim.x + threadIdx.x;   // b*1024 + j
    int b = idx >> 10, j = idx & 1023;
    const float* g = gates + (size_t)b * 4096;
    float ig = sigmoidf_(g[j]);
    float fg = sigmoidf_(g[1024 + j]);
    float gg = tanhf(g[2048 + j]);
    float og = sigmoidf_(g[3072 + j]);
    float cn = fg * c[idx] + ig * gg;
    float hn = og * tanhf(cn);
    c[idx] = cn;
    xh[(size_t)b * 2048 + 1024 + j] = (_Float16)hn;
    out[((size_t)b * T_ + t) * H_ + j] = hn * (float)masks[b * T_ + t];
}

// ---------------------------------------------------------------------------
extern "C" void kernel_launch(void* const* d_in, const int* in_sizes, int n_in,
                              void* d_out, int out_size, void* d_ws, size_t ws_size,
                              hipStream_t stream) {
    (void)in_sizes; (void)n_in; (void)out_size; (void)ws_size;

    const float* features    = (const float*)d_in[0];
    const int*   captions    = (const int*)d_in[1];
    const int*   masks       = (const int*)d_in[2];
    const float* image_att_w = (const float*)d_in[3];
    const float* init_h_w    = (const float*)d_in[4];
    const float* init_h_b    = (const float*)d_in[5];
    const float* init_c_w    = (const float*)d_in[6];
    const float* init_c_b    = (const float*)d_in[7];
    const float* att_hh_w    = (const float*)d_in[8];
    const float* att_hh_b    = (const float*)d_in[9];
    const float* weight_att  = (const float*)d_in[10];
    const float* embedding   = (const float*)d_in[11];
    const float* W_ih        = (const float*)d_in[12];
    const float* W_hh        = (const float*)d_in[13];
    const float* b_ih        = (const float*)d_in[14];
    const float* b_hh        = (const float*)d_in[15];
    float* out = (float*)d_out;

    // ---- workspace carve-up ----
    char* wsp = (char*)d_ws;
    auto alloc = [&](size_t bytes) -> char* {
        char* p = wsp; wsp += (bytes + 255) & ~(size_t)255; return p;
    };
    _Float16* ctx16    = (_Float16*)alloc((size_t)B_ * L_ * D_ * 2);  // 51.4 MB
    _Float16* Wc16     = (_Float16*)alloc((size_t)4 * H_ * 2 * H_);   // 16.8 MB
    _Float16* inith16  = (_Float16*)alloc((size_t)H_ * D_ * 2);
    _Float16* initc16  = (_Float16*)alloc((size_t)H_ * D_ * 2);
    _Float16* atthh16  = (_Float16*)alloc((size_t)D_ * H_ * 2);
    _Float16* imgT16   = (_Float16*)alloc((size_t)D_ * D_ * 2);
    float*    biasc    = (float*)alloc((size_t)4 * H_ * 4);
    float*    fmean    = (float*)alloc((size_t)B_ * D_ * 4);
    float*    h0tmp    = (float*)alloc((size_t)B_ * H_ * 4);
    float*    cbuf     = (float*)alloc((size_t)B_ * H_ * 4);
    float*    hpbuf    = (float*)alloc((size_t)B_ * D_ * 4);
    _Float16* xh       = (_Float16*)alloc((size_t)B_ * 2 * H_ * 2);   // [emb|ctx|h]
    float*    gatesbuf = (float*)alloc((size_t)B_ * 4 * H_ * 4);

    // ---- one-time weight prep ----
    cvt_f32_to_f16<<<(H_ * D_) / 256, 256, 0, stream>>>(init_h_w, inith16, H_ * D_);
    cvt_f32_to_f16<<<(H_ * D_) / 256, 256, 0, stream>>>(init_c_w, initc16, H_ * D_);
    cvt_f32_to_f16<<<(D_ * H_) / 256, 256, 0, stream>>>(att_hh_w, atthh16, D_ * H_);
    transpose_512_f16<<<(D_ * D_) / 256, 256, 0, stream>>>(image_att_w, imgT16);
    fuse_lstm_w<<<(4 * H_ * 2 * H_) / 256, 256, 0, stream>>>(W_ih, W_hh, Wc16);
    fuse_bias<<<(4 * H_) / 256, 256, 0, stream>>>(b_ih, b_hh, biasc);

    // ---- init state ----
    fmean_kernel<<<B_, 256, 0, stream>>>(features, fmean);
    gemm_wmma<true, false><<<dim3(H_ / 128, B_ / 64), 256, 0, stream>>>(
        fmean, D_, inith16, D_, h0tmp, H_, init_h_b, D_);
    gemm_wmma<true, false><<<dim3(H_ / 128, B_ / 64), 256, 0, stream>>>(
        fmean, D_, initc16, D_, cbuf, H_, init_c_b, D_);
    h_to_xh<<<(B_ * H_) / 256, 256, 0, stream>>>(h0tmp, xh);

    // ---- ctx_enc = features @ image_att_w  (fp16 out, features read once) ----
    gemm_wmma<true, true><<<dim3(D_ / 128, (B_ * L_) / 64), 256, 0, stream>>>(
        features, D_, imgT16, D_, ctx16, D_, nullptr, D_);

    // ---- recurrent loop ----
    for (int t = 0; t < T_; t++) {
        // h_proj = h @ att_hh_w.T + b  (A = h slice of xh, lda=2048)
        gemm_wmma<false, false><<<dim3(D_ / 128, B_ / 64), 256, 0, stream>>>(
            xh + 1024, 2 * H_, atthh16, H_, hpbuf, D_, att_hh_b, H_);
        // xh[:,0:512] = embedding[captions[:,t]]
        embed_kernel<<<B_, 256, 0, stream>>>(embedding, captions, t, xh);
        // xh[:,512:1024] = attention context
        attention_kernel<<<B_, 256, 0, stream>>>(ctx16, features, hpbuf,
                                                 weight_att, xh);
        // gates = [emb|ctx|h] @ [W_ih|W_hh].T + (b_ih + b_hh)
        gemm_wmma<false, false><<<dim3((4 * H_) / 128, B_ / 64), 256, 0, stream>>>(
            xh, 2 * H_, Wc16, 2 * H_, gatesbuf, 4 * H_, biasc, 2 * H_);
        // pointwise LSTM + masked output
        lstm_kernel<<<(B_ * H_) / 256, 256, 0, stream>>>(gatesbuf, cbuf, masks,
                                                         t, xh, out);
    }
}
